// CircuitEncoder_26448408609249
// MI455X (gfx1250) — matrix operand order repaired and checked by reading, not compiled, not run on hardware
//
#include <hip/hip_runtime.h>
#include <math.h>

// ---------------------------------------------------------------------------
// CDNA5 / gfx1250 implementation of the two-branch GAT circuit encoder.
// GEMMs: operands pre-packed to zero-padded f16 (A row-major, W transposed),
// then a branch-free v_wmma_f32_16x16x32_f16 inner loop with b128 loads.
// Graph softmax / aggregation: f32 global atomics. LayerNorm: wave32 shuffles.
// ---------------------------------------------------------------------------

typedef __attribute__((ext_vector_type(16))) _Float16 v16h;
typedef __attribute__((ext_vector_type(8)))  _Float16 v8h;
typedef __attribute__((ext_vector_type(8)))  float    v8f;

// float atomic max via integer punning (works for any sign mix, init = -inf)
__device__ __forceinline__ void atomicMaxF(float* addr, float val) {
  if (val >= 0.0f) atomicMax((int*)addr, __float_as_int(val));
  else             atomicMin((unsigned int*)addr, __float_as_uint(val));
}

__global__ void fill_kernel(float* __restrict__ p, float v, long long n) {
  long long i = (long long)blockIdx.x * blockDim.x + threadIdx.x;
  if (i < n) p[i] = v;
}

// A[N,K] f32 -> A16[N,K32] f16 zero-padded
__global__ void pack_a_kernel(const float* __restrict__ A, _Float16* __restrict__ A16,
                              int N, int K, int K32) {
  long long t = (long long)blockIdx.x * blockDim.x + threadIdx.x;
  if (t >= (long long)N * K32) return;
  int k = (int)(t % K32);
  long long n = t / K32;
  A16[t] = (k < K) ? (_Float16)A[n * K + k] : (_Float16)0.0f;
}

// W[K,M] f32 -> Wt16[M,K32] f16 zero-padded (transposed)
__global__ void pack_wt_kernel(const float* __restrict__ W, _Float16* __restrict__ Wt,
                               int K, int M, int K32) {
  long long t = (long long)blockIdx.x * blockDim.x + threadIdx.x;
  if (t >= (long long)M * K32) return;
  int k = (int)(t % K32);
  long long m = t / K32;
  Wt[t] = (k < K) ? (_Float16)W[(long long)k * M + m] : (_Float16)0.0f;
}

// ---------------------------------------------------------------------------
// C[N,M] = A16[N,K32] @ Wt16[M,K32]^T (+ bias[M], optional ReLU)
// One wave per 16x16 C tile; N, M multiples of 16; K32 multiple of 32.
// VGPR layouts per CDNA5 ISA 7.12.2 (wave32):
//   A 16x32 f16 : row = lane&15 ; lane-group selects K-halves {0..7,16..23}/{8..15,24..31}
//   B 32x16 f16 : col = lane&15 ; lane-group selects K-halves {0..7,16..23}/{8..15,24..31}
//   C 16x16 f32 : col = lane&15 ; vgpr r -> row r (lanes<16) / r+8 (lanes>=16)
// Inner loop: 4x global_load_b128 + v_wmma, no guards.
// ---------------------------------------------------------------------------
__global__ void wmma_gemm_kernel(const _Float16* __restrict__ A16,
                                 const _Float16* __restrict__ Wt16,
                                 const float* __restrict__ bias,
                                 float* __restrict__ C,
                                 int N, int K32, int M, int act) {
  int lane = threadIdx.x & 31;
  int gid  = blockIdx.x * (blockDim.x >> 5) + (threadIdx.x >> 5);
  int tiles_m = M >> 4;
  int tiles_n = N >> 4;
  if (gid >= tiles_m * tiles_n) return;
  int tn = gid / tiles_m;
  int tm = gid % tiles_m;

  int half = lane >> 4;     // lane group 0/1
  int l15  = lane & 15;

  const _Float16* Arow = A16 + (long long)(tn * 16 + l15) * K32 + half * 8;
  const _Float16* Brow = Wt16 + (long long)(tm * 16 + l15) * K32 + half * 8;

  v8f acc = {};
#pragma unroll 4
  for (int k0 = 0; k0 < K32; k0 += 32) {
    v8h a0 = *(const v8h*)(Arow + k0);        // K = k0+off+0..7
    v8h a1 = *(const v8h*)(Arow + k0 + 16);   // K = k0+16+off+0..7
    v8h b0 = *(const v8h*)(Brow + k0);        // K = k0+off+0..7
    v8h b1 = *(const v8h*)(Brow + k0 + 16);   // K = k0+16+off+0..7
    v16h a = __builtin_shufflevector(a0, a1, 0,1,2,3,4,5,6,7,8,9,10,11,12,13,14,15);
    v16h b = __builtin_shufflevector(b0, b1, 0,1,2,3,4,5,6,7,8,9,10,11,12,13,14,15);
    acc = __builtin_amdgcn_wmma_f32_16x16x32_f16(false, a, false, b,
                                                 (short)0, acc, false, false);
  }

  int bcol = tm * 16 + l15;
  float bv = bias ? bias[bcol] : 0.0f;
  int mbase = half * 8;
#pragma unroll
  for (int r = 0; r < 8; ++r) {
    float v = acc[r] + bv;
    if (act == 1) v = fmaxf(v, 0.0f);
    C[(long long)(tn * 16 + mbase + r) * M + bcol] = v;
  }
}

// a_src[n,h] = <h[n,h,:], att_src[h,:]> ; same for a_dst
__global__ void att_scores_kernel(const float* __restrict__ h,
                                  const float* __restrict__ att_s,
                                  const float* __restrict__ att_d,
                                  float* __restrict__ asrc, float* __restrict__ adst,
                                  int N, int H, int C) {
  int t = blockIdx.x * blockDim.x + threadIdx.x;
  if (t >= N * H) return;
  int n = t / H, hd = t % H;
  const float* hp = h + (long long)n * H * C + (long long)hd * C;
  const float* as = att_s + hd * C;
  const float* ad = att_d + hd * C;
  float s = 0.f, d = 0.f;
  for (int c = 0; c < C; ++c) { s += hp[c] * as[c]; d += hp[c] * ad[c]; }
  asrc[t] = s; adst[t] = d;
}

// edge idx in [0, E+N): idx>=E are self loops. leaky_relu logits + segment max.
__global__ void edge_logits_max_kernel(const int* __restrict__ src, const int* __restrict__ dst,
                                       int E, int N, int H,
                                       const float* __restrict__ asrc,
                                       const float* __restrict__ adst,
                                       float* __restrict__ att_e, float* __restrict__ amax) {
  int idx = blockIdx.x * blockDim.x + threadIdx.x;
  if (idx >= E + N) return;
  int s = (idx < E) ? src[idx] : (idx - E);
  int d = (idx < E) ? dst[idx] : (idx - E);
  for (int hd = 0; hd < H; ++hd) {
    float a = asrc[s * H + hd] + adst[d * H + hd];
    a = (a > 0.f) ? a : 0.2f * a;                // LeakyReLU(0.2)
    att_e[(long long)idx * H + hd] = a;
    atomicMaxF(&amax[d * H + hd], a);
  }
}

__global__ void edge_exp_sum_kernel(const int* __restrict__ dst, int E, int N, int H,
                                    float* __restrict__ att_e,
                                    const float* __restrict__ amax,
                                    float* __restrict__ denom) {
  int idx = blockIdx.x * blockDim.x + threadIdx.x;
  if (idx >= E + N) return;
  int d = (idx < E) ? dst[idx] : (idx - E);
  for (int hd = 0; hd < H; ++hd) {
    float e = expf(att_e[(long long)idx * H + hd] - amax[d * H + hd]);
    att_e[(long long)idx * H + hd] = e;
    atomicAdd(&denom[d * H + hd], e);
  }
}

__global__ void edge_div_kernel(const int* __restrict__ dst, int E, int N, int H,
                                float* __restrict__ att_e, const float* __restrict__ denom) {
  int t = blockIdx.x * blockDim.x + threadIdx.x;
  if (t >= (E + N) * H) return;
  int idx = t / H, hd = t % H;
  int d = (idx < E) ? dst[idx] : (idx - E);
  att_e[t] = att_e[t] / denom[d * H + hd];
}

// out[d,h,c] += alpha[idx,h] * h[s,h,c]   (grid-stride, coalesced over c)
__global__ void aggregate_kernel(const int* __restrict__ src, const int* __restrict__ dst,
                                 int E, int N, int H, int C,
                                 const float* __restrict__ att_e,
                                 const float* __restrict__ h,
                                 float* __restrict__ out) {
  int HC = H * C;
  long long total = (long long)(E + N) * HC;
  for (long long t = (long long)blockIdx.x * blockDim.x + threadIdx.x; t < total;
       t += (long long)gridDim.x * blockDim.x) {
    int idx = (int)(t / HC);
    int rem = (int)(t % HC);
    int hd = rem / C;
    int s = (idx < E) ? src[idx] : (idx - E);
    int d = (idx < E) ? dst[idx] : (idx - E);
    float alpha = att_e[(long long)idx * H + hd];
    float val = alpha * h[(long long)s * HC + rem];
    atomicAdd(&out[(long long)d * HC + rem], val);
  }
}

// per-node: head concat/mean + bias, LayerNorm, optional ELU. One wave32/node.
__global__ void finalize_kernel(const float* __restrict__ outacc,
                                const float* __restrict__ bias,
                                const float* __restrict__ g, const float* __restrict__ bln,
                                float* __restrict__ xout,
                                int N, int H, int C, int concat, int elu) {
  int node = blockIdx.x * (blockDim.x >> 5) + (threadIdx.x >> 5);
  int lane = threadIdx.x & 31;
  if (node >= N) return;
  int D = concat ? H * C : C;
  int per = (D + 31) >> 5;          // <= 8 for D<=256
  float vals[8];
  float sum = 0.f;
#pragma unroll
  for (int j = 0; j < 8; ++j) {
    if (j >= per) break;
    int c = lane + j * 32;
    float v = 0.f;
    if (c < D) {
      if (concat) {
        v = outacc[(long long)node * H * C + c];
      } else {
        for (int hd = 0; hd < H; ++hd) v += outacc[(long long)node * H * C + hd * C + c];
        v /= (float)H;
      }
      v += bias[c];
      sum += v;
    }
    vals[j] = v;
  }
  for (int m = 16; m >= 1; m >>= 1) sum += __shfl_xor(sum, m, 32);
  float mu = sum / (float)D;
  float var = 0.f;
#pragma unroll
  for (int j = 0; j < 8; ++j) {
    if (j >= per) break;
    int c = lane + j * 32;
    if (c < D) { float dv = vals[j] - mu; var += dv * dv; }
  }
  for (int m = 16; m >= 1; m >>= 1) var += __shfl_xor(var, m, 32);
  float inv = rsqrtf(var / (float)D + 1e-5f);
#pragma unroll
  for (int j = 0; j < 8; ++j) {
    if (j >= per) break;
    int c = lane + j * 32;
    if (c < D) {
      float y = (vals[j] - mu) * inv * g[c] + bln[c];
      if (elu) y = (y > 0.f) ? y : (expf(y) - 1.f);
      xout[(long long)node * D + c] = y;
    }
  }
}

__global__ void pool_accum_kernel(const float* __restrict__ x, const int* __restrict__ batch,
                                  int N, int D, float* __restrict__ sums,
                                  float* __restrict__ cnt) {
  int t = blockIdx.x * blockDim.x + threadIdx.x;
  if (t >= N * D) return;
  int n = t / D, c = t % D;
  int b = batch[n];
  atomicAdd(&sums[b * D + c], x[t]);
  if (c == 0) atomicAdd(&cnt[b], 1.0f);
}

__global__ void pool_write_kernel(const float* __restrict__ sums, const float* __restrict__ cnt,
                                  float* __restrict__ fused, int B, int D,
                                  int col_off, int stride) {
  int t = blockIdx.x * blockDim.x + threadIdx.x;
  if (t >= B * D) return;
  int b = t / D, c = t % D;
  fused[b * stride + col_off + c] = sums[t] / fmaxf(cnt[b], 1.0f);
}

// ---------------------------------------------------------------------------
// Host orchestration
// ---------------------------------------------------------------------------
static inline int cdiv(long long a, int b) { return (int)((a + b - 1) / b); }
static inline int k32of(int K) { return ((K + 31) / 32) * 32; }

struct GatBufs {
  float *hbuf, *obuf, *asrc, *adst, *att_e, *amax, *denom;
  _Float16 *a16, *wt16;
};

// full WMMA GEMM pipeline: pack A + pack W^T + gemm
static void run_gemm(const float* A, const float* W, const float* bias, float* C,
                     int N, int K, int M, int act,
                     _Float16* a16, _Float16* wt16, hipStream_t st) {
  int K32 = k32of(K);
  pack_a_kernel<<<cdiv((long long)N * K32, 256), 256, 0, st>>>(A, a16, N, K, K32);
  pack_wt_kernel<<<cdiv((long long)M * K32, 256), 256, 0, st>>>(W, wt16, K, M, K32);
  int tiles = (M >> 4) * (N >> 4);
  wmma_gemm_kernel<<<cdiv(tiles, 8), 256, 0, st>>>(a16, wt16, bias, C, N, K32, M, act);
}

static void run_gat_layer(const float* x, int N, int K,
                          const int* src, const int* dst, int E,
                          const float* W, const float* att_s, const float* att_d,
                          const float* bias, const float* ln_g, const float* ln_b,
                          int H, int C, int concat, int elu,
                          const GatBufs& b, float* x_out, hipStream_t st) {
  int M = H * C;
  // 1) h = x @ W  (WMMA)
  run_gemm(x, W, nullptr, b.hbuf, N, K, M, 0, b.a16, b.wt16, st);
  // 2) attention scores
  att_scores_kernel<<<cdiv((long long)N * H, 256), 256, 0, st>>>(b.hbuf, att_s, att_d,
                                                                 b.asrc, b.adst, N, H, C);
  // 3) init segment buffers
  fill_kernel<<<cdiv((long long)N * H, 256), 256, 0, st>>>(b.amax, -INFINITY, (long long)N * H);
  fill_kernel<<<cdiv((long long)N * H, 256), 256, 0, st>>>(b.denom, 0.0f, (long long)N * H);
  fill_kernel<<<cdiv((long long)N * M, 256), 256, 0, st>>>(b.obuf, 0.0f, (long long)N * M);
  // 4) logits + segment max
  edge_logits_max_kernel<<<cdiv(E + N, 256), 256, 0, st>>>(src, dst, E, N, H,
                                                           b.asrc, b.adst, b.att_e, b.amax);
  // 5) exp + segment sum
  edge_exp_sum_kernel<<<cdiv(E + N, 256), 256, 0, st>>>(dst, E, N, H, b.att_e, b.amax, b.denom);
  // 6) alpha = e / denom
  edge_div_kernel<<<cdiv((long long)(E + N) * H, 256), 256, 0, st>>>(dst, E, N, H,
                                                                     b.att_e, b.denom);
  // 7) weighted scatter-add aggregation
  aggregate_kernel<<<4096, 256, 0, st>>>(src, dst, E, N, H, C, b.att_e, b.hbuf, b.obuf);
  // 8) concat/mean + bias + LayerNorm (+ELU)
  finalize_kernel<<<cdiv(N, 8), 256, 0, st>>>(b.obuf, bias, ln_g, ln_b, x_out,
                                              N, H, C, concat, elu);
}

extern "C" void kernel_launch(void* const* d_in, const int* in_sizes, int n_in,
                              void* d_out, int out_size, void* d_ws, size_t ws_size,
                              hipStream_t stream) {
  const int N_DAG = 100000, E_DAG = 400000;
  const int N_CPL = 50000,  E_CPL = 200000;
  const int B = 256;

  // ---- inputs (setup_inputs dict order; params = JAX pytree, dict keys sorted)
  const float* dag_x   = (const float*)d_in[0];
  const int*   dag_ei  = (const int*)d_in[1];
  const int*   dag_bat = (const int*)d_in[2];
  const float* coup_x  = (const float*)d_in[3];
  const int*   cpl_ei  = (const int*)d_in[4];
  const int*   cpl_bat = (const int*)d_in[5];
  // params: coup{convs[{W,att_dst,att_src,bias}x2], norms[{b,g}x2]},
  //         dag{convs[...x3], norms[{b,g}x3]}, fusion{W1,W2,b1,b2}
  const float* cW0 = (const float*)d_in[6];
  const float* cAd0= (const float*)d_in[7];
  const float* cAs0= (const float*)d_in[8];
  const float* cB0 = (const float*)d_in[9];
  const float* cW1 = (const float*)d_in[10];
  const float* cAd1= (const float*)d_in[11];
  const float* cAs1= (const float*)d_in[12];
  const float* cB1 = (const float*)d_in[13];
  const float* cNb0= (const float*)d_in[14];
  const float* cNg0= (const float*)d_in[15];
  const float* cNb1= (const float*)d_in[16];
  const float* cNg1= (const float*)d_in[17];
  const float* dW0 = (const float*)d_in[18];
  const float* dAd0= (const float*)d_in[19];
  const float* dAs0= (const float*)d_in[20];
  const float* dB0 = (const float*)d_in[21];
  const float* dW1 = (const float*)d_in[22];
  const float* dAd1= (const float*)d_in[23];
  const float* dAs1= (const float*)d_in[24];
  const float* dB1 = (const float*)d_in[25];
  const float* dW2 = (const float*)d_in[26];
  const float* dAd2= (const float*)d_in[27];
  const float* dAs2= (const float*)d_in[28];
  const float* dB2 = (const float*)d_in[29];
  const float* dNb0= (const float*)d_in[30];
  const float* dNg0= (const float*)d_in[31];
  const float* dNb1= (const float*)d_in[32];
  const float* dNg1= (const float*)d_in[33];
  const float* dNb2= (const float*)d_in[34];
  const float* dNg2= (const float*)d_in[35];
  const float* fW1 = (const float*)d_in[36];
  const float* fW2 = (const float*)d_in[37];
  const float* fb1 = (const float*)d_in[38];
  const float* fb2 = (const float*)d_in[39];

  // ---- workspace carve-up (64-float = 256B aligned slices)
  float* ws = (float*)d_ws;
  long long off = 0;
  auto carve = [&](long long nfloats) -> float* {
    float* p = ws + off;
    off += (nfloats + 63) & ~63LL;
    return p;
  };
  float* xbuf  = carve((long long)N_DAG * 256);   // node features
  float* hbuf  = carve((long long)N_DAG * 256);   // x @ W
  float* obuf  = carve((long long)N_DAG * 256);   // aggregation
  float* asrc  = carve((long long)N_DAG * 4);
  float* adst  = carve((long long)N_DAG * 4);
  float* att_e = carve((long long)(E_DAG + N_DAG) * 4);
  float* amax  = carve((long long)N_DAG * 4);
  float* denom = carve((long long)N_DAG * 4);
  float* psum  = carve((long long)B * 32);
  float* pcnt  = carve(B);
  float* fused = carve((long long)B * 64);
  float* h1    = carve((long long)B * 64);
  _Float16* a16  = (_Float16*)carve((long long)N_DAG * 256 / 2);  // N*K32 halfs max
  _Float16* wt16 = (_Float16*)carve((long long)256 * 256 / 2);    // M*K32 halfs max

  GatBufs gb = { hbuf, obuf, asrc, adst, att_e, amax, denom, a16, wt16 };

  const int* dag_src = dag_ei;
  const int* dag_dst = dag_ei + E_DAG;
  const int* cpl_src = cpl_ei;
  const int* cpl_dst = cpl_ei + E_CPL;

  // ================= DAG encoder =================
  // L1: 20 -> 4x64 concat, LN, ELU
  run_gat_layer(dag_x, N_DAG, 20, dag_src, dag_dst, E_DAG,
                dW0, dAs0, dAd0, dB0, dNg0, dNb0, 4, 64, 1, 1, gb, xbuf, stream);
  // L2: 256 -> 4x64 concat, LN, ELU
  run_gat_layer(xbuf, N_DAG, 256, dag_src, dag_dst, E_DAG,
                dW1, dAs1, dAd1, dB1, dNg1, dNb1, 4, 64, 1, 1, gb, xbuf, stream);
  // L3: 256 -> 1x32 mean, LN (no ELU)
  run_gat_layer(xbuf, N_DAG, 256, dag_src, dag_dst, E_DAG,
                dW2, dAs2, dAd2, dB2, dNg2, dNb2, 1, 32, 0, 0, gb, xbuf, stream);
  // mean pool -> fused[:, 0:32]
  fill_kernel<<<cdiv(B * 32, 256), 256, 0, stream>>>(psum, 0.0f, (long long)B * 32);
  fill_kernel<<<1, 256, 0, stream>>>(pcnt, 0.0f, B);
  pool_accum_kernel<<<cdiv((long long)N_DAG * 32, 256), 256, 0, stream>>>(xbuf, dag_bat,
                                                                          N_DAG, 32, psum, pcnt);
  pool_write_kernel<<<cdiv(B * 32, 256), 256, 0, stream>>>(psum, pcnt, fused, B, 32, 0, 64);

  // ================= Coupling encoder =================
  // L1: 4 -> 4x32 concat, LN, ELU
  run_gat_layer(coup_x, N_CPL, 4, cpl_src, cpl_dst, E_CPL,
                cW0, cAs0, cAd0, cB0, cNg0, cNb0, 4, 32, 1, 1, gb, xbuf, stream);
  // L2: 128 -> 1x32 mean, LN (no ELU)
  run_gat_layer(xbuf, N_CPL, 128, cpl_src, cpl_dst, E_CPL,
                cW1, cAs1, cAd1, cB1, cNg1, cNb1, 1, 32, 0, 0, gb, xbuf, stream);
  // mean pool -> fused[:, 32:64]
  fill_kernel<<<cdiv(B * 32, 256), 256, 0, stream>>>(psum, 0.0f, (long long)B * 32);
  fill_kernel<<<1, 256, 0, stream>>>(pcnt, 0.0f, B);
  pool_accum_kernel<<<cdiv((long long)N_CPL * 32, 256), 256, 0, stream>>>(xbuf, cpl_bat,
                                                                          N_CPL, 32, psum, pcnt);
  pool_write_kernel<<<cdiv(B * 32, 256), 256, 0, stream>>>(psum, pcnt, fused, B, 32, 32, 64);

  // ================= Fusion MLP (WMMA) =================
  run_gemm(fused, fW1, fb1, h1, B, 64, 64, 1, a16, wt16, stream);
  run_gemm(h1, fW2, fb2, (float*)d_out, B, 64, 64, 0, a16, wt16, stream);

  (void)in_sizes; (void)n_in; (void)out_size; (void)ws_size;
}
